// EmbeddingBlock_36352603194139
// MI455X (gfx1250) — matrix operand order repaired and checked by reading, not compiled, not run on hardware
//
#include <hip/hip_runtime.h>
#include <hip/hip_bf16.h>

typedef _Float16 v16h __attribute__((ext_vector_type(16)));
typedef _Float16 v8h  __attribute__((ext_vector_type(8)));
typedef _Float16 v2h  __attribute__((ext_vector_type(2)));
typedef float    v8f  __attribute__((ext_vector_type(8)));

#define N_NODES    50000
#define N_EDGES    800000
#define NUM_RADIAL 6
#define HIDDEN     128
#define K3         384   // 3*HIDDEN
#define KT_TOTAL   12    // 384/32
#define NT_TOTAL   8     // 128/16

// ---------------------------------------------------------------------------
// Kernel 1: repack W_lin [128,384] fp32 into WMMA B-fragment layout (f16).
// Layout: Bpack[kt][nt][lane][p], p=0..15 halves per lane.
//   lane L holds column N = nt*16 + (L%16),
//   half p holds K = kt*32 + (p&7) + (L>=16 ? 8 : 0) + (p>=8 ? 16 : 0)
// so each lane's 16 halves are one contiguous 32B vector at load time.
// ---------------------------------------------------------------------------
__global__ void pack_weights_kernel(const float* __restrict__ W_lin,
                                    _Float16* __restrict__ Bpack) {
    int idx = blockIdx.x * blockDim.x + threadIdx.x;
    if (idx >= KT_TOTAL * NT_TOTAL * 32 * 16) return;
    int p    = idx & 15;
    int lane = (idx >> 4) & 31;
    int nt   = (idx >> 9) & 7;
    int kt   = idx >> 12;
    int n = nt * 16 + (lane & 15);
    int k = kt * 32 + (p & 7) + ((lane >= 16) ? 8 : 0) + ((p >= 8) ? 16 : 0);
    Bpack[idx] = (_Float16)W_lin[n * K3 + k];
}

// ---------------------------------------------------------------------------
// Kernel 2: node features = concat(resi_emb[x0], atom_emb[x1]) -> f16 [N,128]
// ---------------------------------------------------------------------------
__global__ void node_feat_kernel(const int* __restrict__ x,
                                 const float* __restrict__ resi_w,
                                 const float* __restrict__ atom_w,
                                 _Float16* __restrict__ node_f16) {
    int idx = blockIdx.x * blockDim.x + threadIdx.x;
    if (idx >= N_NODES * HIDDEN) return;
    int n = idx >> 7;
    int k = idx & 127;
    float v = (k < 64) ? resi_w[x[2 * n + 0] * 64 + k]
                       : atom_w[x[2 * n + 1] * 64 + (k - 64)];
    node_f16[idx] = (_Float16)v;
}

// ---------------------------------------------------------------------------
// Kernel 3: each wave owns TWO 16-edge tiles (32 edges) so every B fragment
// feeds two WMMAs (load:wmma ratio 1.25 instead of 2.25).
//   - rbf MLP (swish) -> per-wave LDS tile [32][128] f16
//   - [32,384] x [384,128] via 192x v_wmma_f32_16x16x32_f16
//   - +bias, swish, fp32 store
// 4 waves per block (128 threads), static LDS ~36KB.
// ---------------------------------------------------------------------------
__global__ __launch_bounds__(128)
void embed_gemm_kernel(const float* __restrict__ rbf,
                       const int*   __restrict__ ei,
                       const int*   __restrict__ ej,
                       const float* __restrict__ W_rbf,
                       const float* __restrict__ b_rbf,
                       const _Float16* __restrict__ node_f16,
                       const _Float16* __restrict__ Bpack,
                       const float* __restrict__ b_lin,
                       float* __restrict__ out) {
    __shared__ float s_wrbf[HIDDEN * NUM_RADIAL];
    __shared__ float s_brbf[HIDDEN];
    __shared__ alignas(32) _Float16 s_rbfh[4][32 * HIDDEN];

    const int tid = threadIdx.x;
    for (int t = tid; t < HIDDEN * NUM_RADIAL; t += 128) s_wrbf[t] = W_rbf[t];
    if (tid < HIDDEN) s_brbf[tid] = b_rbf[tid];
    __syncthreads();

    const int wave  = tid >> 5;
    const int lane  = tid & 31;
    const int r     = lane & 15;   // edge row within a 16-edge tile (M)
    const int hi    = lane >> 4;   // half-wave select
    const int pair  = blockIdx.x * 4 + wave;   // index of 32-edge super-tile
    const int e0    = pair * 32 + r;           // tile-0 edge for this lane
    const int e1    = e0 + 16;                 // tile-1 edge for this lane
    const int base0 = hi * 8;

    // ---- rbf projection + swish into this wave's LDS tile [32][128] f16 ----
    const int fbase = hi * 64;   // lanes 0-15 do features 0..63, 16-31 do 64..127
#pragma unroll
    for (int t = 0; t < 2; ++t) {
        const int e = (t == 0) ? e0 : e1;
        float rv[NUM_RADIAL];
#pragma unroll
        for (int q = 0; q < NUM_RADIAL; ++q) rv[q] = rbf[e * NUM_RADIAL + q];
        _Float16* row = &s_rbfh[wave][(t * 16 + r) * HIDDEN];
        for (int c = 0; c < 64; c += 2) {
            float a0 = s_brbf[fbase + c];
            float a1 = s_brbf[fbase + c + 1];
#pragma unroll
            for (int q = 0; q < NUM_RADIAL; ++q) {
                a0 += rv[q] * s_wrbf[(fbase + c)     * NUM_RADIAL + q];
                a1 += rv[q] * s_wrbf[(fbase + c + 1) * NUM_RADIAL + q];
            }
            a0 = a0 / (1.0f + __expf(-a0));   // swish
            a1 = a1 / (1.0f + __expf(-a1));
            v2h pk; pk.x = (_Float16)a0; pk.y = (_Float16)a1;
            *(v2h*)(row + fbase + c) = pk;
        }
    }

    // ---- GEMM: C[32x128] += A[32x384] * B[384x128] ----
    const _Float16* rowI0 = node_f16 + (size_t)ei[e0] * HIDDEN;
    const _Float16* rowJ0 = node_f16 + (size_t)ej[e0] * HIDDEN;
    const _Float16* rowI1 = node_f16 + (size_t)ei[e1] * HIDDEN;
    const _Float16* rowJ1 = node_f16 + (size_t)ej[e1] * HIDDEN;
    const v16h* Bv = (const v16h*)Bpack;   // index = (kt*8+nt)*32 + lane

    v8f acc0[NT_TOTAL] = {};
    v8f acc1[NT_TOTAL] = {};

    // K-tiles 0..7: node[i] (kt 0..3) and node[j] (kt 4..7), from global f16
#pragma unroll
    for (int kt = 0; kt < 8; ++kt) {
        const _Float16* s0 = (kt < 4) ? rowI0 : rowJ0;
        const _Float16* s1 = (kt < 4) ? rowI1 : rowJ1;
        const int koff = (kt & 3) * 32 + base0;
        v8h lo0 = *(const v8h*)(s0 + koff);
        v8h hi0 = *(const v8h*)(s0 + koff + 16);
        v8h lo1 = *(const v8h*)(s1 + koff);
        v8h hi1 = *(const v8h*)(s1 + koff + 16);
        v16h a0 = __builtin_shufflevector(lo0, hi0, 0, 1, 2, 3, 4, 5, 6, 7,
                                                    8, 9, 10, 11, 12, 13, 14, 15);
        v16h a1 = __builtin_shufflevector(lo1, hi1, 0, 1, 2, 3, 4, 5, 6, 7,
                                                    8, 9, 10, 11, 12, 13, 14, 15);
#pragma unroll
        for (int nt = 0; nt < NT_TOTAL; ++nt) {
            v16h b = Bv[(kt * NT_TOTAL + nt) * 32 + lane];
            acc0[nt] = __builtin_amdgcn_wmma_f32_16x16x32_f16(
                false, a0, false, b, (short)0, acc0[nt], false, false);
            acc1[nt] = __builtin_amdgcn_wmma_f32_16x16x32_f16(
                false, a1, false, b, (short)0, acc1[nt], false, false);
        }
    }

    // K-tiles 8..11: rbf_h tile from LDS
    const _Float16* lrow0 = &s_rbfh[wave][r * HIDDEN];
    const _Float16* lrow1 = &s_rbfh[wave][(16 + r) * HIDDEN];
#pragma unroll
    for (int kt = 0; kt < 4; ++kt) {
        const int koff = kt * 32 + base0;
        v8h lo0 = *(const v8h*)(lrow0 + koff);
        v8h hi0 = *(const v8h*)(lrow0 + koff + 16);
        v8h lo1 = *(const v8h*)(lrow1 + koff);
        v8h hi1 = *(const v8h*)(lrow1 + koff + 16);
        v16h a0 = __builtin_shufflevector(lo0, hi0, 0, 1, 2, 3, 4, 5, 6, 7,
                                                    8, 9, 10, 11, 12, 13, 14, 15);
        v16h a1 = __builtin_shufflevector(lo1, hi1, 0, 1, 2, 3, 4, 5, 6, 7,
                                                    8, 9, 10, 11, 12, 13, 14, 15);
#pragma unroll
        for (int nt = 0; nt < NT_TOTAL; ++nt) {
            v16h b = Bv[((kt + 8) * NT_TOTAL + nt) * 32 + lane];
            acc0[nt] = __builtin_amdgcn_wmma_f32_16x16x32_f16(
                false, a0, false, b, (short)0, acc0[nt], false, false);
            acc1[nt] = __builtin_amdgcn_wmma_f32_16x16x32_f16(
                false, a1, false, b, (short)0, acc1[nt], false, false);
        }
    }

    // ---- epilogue: +bias, swish, store fp32 ----
    // C layout: VGPR v, lane L -> row m = v + (L>=16?8:0), col n = nt*16 + L%16
    const int mBase = hi * 8;
    const size_t row0 = (size_t)(pair * 32) * HIDDEN;        // tile 0
    const size_t row1 = (size_t)(pair * 32 + 16) * HIDDEN;   // tile 1
#pragma unroll
    for (int nt = 0; nt < NT_TOTAL; ++nt) {
        const int n = nt * 16 + r;
        const float bl = b_lin[n];
#pragma unroll
        for (int v = 0; v < 8; ++v) {
            float v0 = acc0[nt][v] + bl;
            float v1 = acc1[nt][v] + bl;
            v0 = v0 / (1.0f + __expf(-v0));   // swish
            v1 = v1 / (1.0f + __expf(-v1));
            out[row0 + (size_t)(v + mBase) * HIDDEN + n] = v0;
            out[row1 + (size_t)(v + mBase) * HIDDEN + n] = v1;
        }
    }
}

// ---------------------------------------------------------------------------
extern "C" void kernel_launch(void* const* d_in, const int* in_sizes, int n_in,
                              void* d_out, int out_size, void* d_ws, size_t ws_size,
                              hipStream_t stream) {
    const int*   x     = (const int*)d_in[0];
    const float* rbf   = (const float*)d_in[1];
    const int*   ei    = (const int*)d_in[2];
    const int*   ej    = (const int*)d_in[3];
    const float* resiW = (const float*)d_in[4];
    const float* atomW = (const float*)d_in[5];
    const float* Wrbf  = (const float*)d_in[6];
    const float* brbf  = (const float*)d_in[7];
    const float* Wlin  = (const float*)d_in[8];
    const float* blin  = (const float*)d_in[9];
    float* out = (float*)d_out;

    _Float16* Bpack = (_Float16*)d_ws;                       // 96 KB
    _Float16* nodeF = (_Float16*)((char*)d_ws + 131072);     // 12.8 MB

    const int packN = KT_TOTAL * NT_TOTAL * 32 * 16;         // 49152
    pack_weights_kernel<<<(packN + 255) / 256, 256, 0, stream>>>(Wlin, Bpack);

    const int nodeN = N_NODES * HIDDEN;                      // 6.4M
    node_feat_kernel<<<(nodeN + 255) / 256, 256, 0, stream>>>(x, resiW, atomW, nodeF);

    // 800000 edges / 32 per wave / 4 waves per block = 6250 blocks, 128 threads
    embed_gemm_kernel<<<N_EDGES / 32 / 4, 128, 0, stream>>>(
        rbf, ei, ej, Wrbf, brbf, nodeF, Bpack, blin, out);
}
